// Mamba3Layer_39298950758436
// MI455X (gfx1250) — matrix operand order repaired and checked by reading, not compiled, not run on hardware
//
#include <hip/hip_runtime.h>
#include <cstdint>
#include <cstddef>

// ------------------------- problem constants -------------------------
constexpr int kDModel  = 2048;
constexpr int kDState  = 64;
constexpr int kNHeads  = 64;
constexpr int kChunk   = 256;
constexpr int kDInner  = 4096;
constexpr int kHeadDim = 64;
constexpr int kBC      = 64;
constexpr int kDProj   = 2 * kDInner + 2 * kBC + 2 * kNHeads + kDState / 2; // 8480
constexpr int kBatch   = 2;
constexpr int kT       = 2048;
constexpr int kM       = kBatch * kT;     // 4096 (rows of big GEMMs)
constexpr int kNChunks = kT / kChunk;     // 8
constexpr float kScale = 0.125f;          // 64^-0.5
constexpr float kEps   = 1e-6f;

// proj row column offsets
constexpr int ZOFF  = 0;
constexpr int XOFF  = kDInner;              // 4096
constexpr int BOFF  = 2 * kDInner;          // 8192
constexpr int COFF  = 2 * kDInner + kBC;    // 8256
constexpr int DTOFF = 2 * kDInner + 2*kBC;  // 8320
constexpr int LOFF  = DTOFF + kNHeads;      // 8384
constexpr int THOFF = LOFF + kNHeads;       // 8448

// ------------------------- types / helpers -------------------------
typedef __bf16 bf16_t;
typedef __attribute__((ext_vector_type(16))) __bf16 v16bf;
typedef __attribute__((ext_vector_type(8)))  float  v8f;
typedef __attribute__((ext_vector_type(4)))  uint32_t su4;
typedef __attribute__((ext_vector_type(8)))  uint32_t su8;

__device__ __forceinline__ bf16_t f2bf(float f) {
  union { float f; uint32_t u; } v; v.f = f;
  uint32_t r = v.u + 0x7FFFu + ((v.u >> 16) & 1u); // round-to-nearest-even
  uint16_t h = (uint16_t)(r >> 16);
  bf16_t out; __builtin_memcpy(&out, &h, 2);
  return out;
}
__device__ __forceinline__ float bf2f(bf16_t b) {
  uint16_t h; __builtin_memcpy(&h, &b, 2);
  union { uint32_t u; float f; } v; v.u = ((uint32_t)h) << 16;
  return v.f;
}
__device__ __forceinline__ uint32_t pack2(float a, float b) {
  bf16_t ba = f2bf(a), bb = f2bf(b);
  uint16_t ha, hb;
  __builtin_memcpy(&ha, &ba, 2);
  __builtin_memcpy(&hb, &bb, 2);
  return (uint32_t)ha | ((uint32_t)hb << 16);
}
__device__ __forceinline__ float siluf(float x) { return x / (1.f + __expf(-x)); }

__device__ __forceinline__ v8f wmma_bf16(v16bf a, v16bf b, v8f c) {
  return __builtin_amdgcn_wmma_f32_16x16x32_bf16(false, a, false, b, (short)0, c,
                                                 false, false);
}

// LDS byte address of a shared-memory pointer (flat low 32 bits).
__device__ __forceinline__ uint32_t lds_off(const void* p) {
  return (uint32_t)(uintptr_t)p;
}

// CDNA5 async global->LDS copy, 16B per lane, tracked by ASYNCcnt.
__device__ __forceinline__ void g2l_async_b128(uint32_t lds_byte_addr, const void* g) {
  asm volatile("global_load_async_to_lds_b128 %0, %1, off"
               :: "v"(lds_byte_addr), "v"(g)
               : "memory");
}
__device__ __forceinline__ void wait_async0() {
#if __has_builtin(__builtin_amdgcn_s_wait_asynccnt)
  __builtin_amdgcn_s_wait_asynccnt(0);
#else
  asm volatile("s_wait_asynccnt 0x0" ::: "memory");
#endif
}
__device__ __forceinline__ void wait_tensor0() {
#if __has_builtin(__builtin_amdgcn_s_wait_tensorcnt)
  __builtin_amdgcn_s_wait_tensorcnt(0);
#else
  asm volatile("s_wait_tensorcnt 0x0" ::: "memory");
#endif
}

// Tensor Data Mover: contiguous 1-D tile (nelem bf16 elements) global->LDS.
// Builds D# group0 (4 SGPRs) + group1 (8 SGPRs) per CDNA5 ISA ch.8; issued
// once per wave (EXEC is ignored by TDM), drained with s_wait_tensorcnt.
__device__ __forceinline__ void tdm_load_1d(uint32_t lds_addr, const void* gptr,
                                            uint32_t nelem) {
  uint64_t ga = (uint64_t)(uintptr_t)gptr;
  su4 g0;
  g0.x = (uint32_t)__builtin_amdgcn_readfirstlane((int)1u);  // count=1, user mode
  g0.y = (uint32_t)__builtin_amdgcn_readfirstlane((int)lds_addr);
  g0.z = (uint32_t)__builtin_amdgcn_readfirstlane((int)(uint32_t)ga);
  g0.w = (uint32_t)__builtin_amdgcn_readfirstlane(
      (int)(((uint32_t)(ga >> 32) & 0x01FFFFFFu) | (2u << 30)));  // type=2
  su8 g1;
  g1[0] = (uint32_t)__builtin_amdgcn_readfirstlane((int)(1u << 16)); // data_size=2B
  g1[1] = (uint32_t)__builtin_amdgcn_readfirstlane((int)(nelem << 16));      // dim0 lo16
  g1[2] = (uint32_t)__builtin_amdgcn_readfirstlane((int)((nelem >> 16) | (1u << 16))); // dim0 hi16, dim1=1
  g1[3] = (uint32_t)__builtin_amdgcn_readfirstlane((int)(nelem << 16));      // tile_dim0
  g1[4] = (uint32_t)__builtin_amdgcn_readfirstlane((int)1u);                 // tile_dim1=1
  g1[5] = (uint32_t)__builtin_amdgcn_readfirstlane((int)nelem);              // dim0 stride lo
  g1[6] = (uint32_t)__builtin_amdgcn_readfirstlane((int)0u);
  g1[7] = (uint32_t)__builtin_amdgcn_readfirstlane((int)0u);
  asm volatile("tensor_load_to_lds %0, %1" :: "s"(g0), "s"(g1) : "memory");
}

// ---- CDNA5 WMMA fragment loads, fully vectorized (2 x b128 each) ----
__device__ __forceinline__ v16bf ldsA(const bf16_t* src, int ld, int row0, int k0, int lane) {
  int ll = lane & 15, hf = lane >> 4;
  const char* base = (const char*)(src + (size_t)(row0 + ll) * ld + k0 + 16 * hf);
  union { uint4 u[2]; v16bf b; } cvt;
  cvt.u[0] = *(const uint4*)(base);
  cvt.u[1] = *(const uint4*)(base + 16);   // +8 bf16 in K
  return cvt.b;
}
__device__ __forceinline__ v16bf ldsBT(const bf16_t* src, int ld, int k0, int col0, int lane) {
  int ll = lane & 15, hf = lane >> 4;
  const char* base = (const char*)(src + (size_t)(col0 + ll) * ld + k0 + 16 * hf);
  union { uint4 u[2]; v16bf b; } cvt;
  cvt.u[0] = *(const uint4*)(base);
  cvt.u[1] = *(const uint4*)(base + 16);
  return cvt.b;
}

// ------------------------- kernels -------------------------

__global__ void k_f2bf(const float* __restrict__ src, bf16_t* __restrict__ dst, size_t n) {
  size_t i = (size_t)blockIdx.x * blockDim.x + threadIdx.x;
  if (i < n) dst[i] = f2bf(src[i]);
}

// fp32 [K][N] -> bf16 transposed [N][K], 32x32 LDS tiles (for weights).
__global__ __launch_bounds__(256) void k_f2bfT(const float* __restrict__ src,
                                               bf16_t* __restrict__ dst,
                                               int K, int N) {
  __shared__ float tile[32][33];
  int n0 = blockIdx.x * 32;
  int k0 = blockIdx.y * 32;
  int c  = threadIdx.x & 31;
  int r8 = threadIdx.x >> 5;     // 0..7
#pragma unroll
  for (int q = 0; q < 4; ++q) {
    int r = r8 + q * 8;
    tile[r][c] = src[(size_t)(k0 + r) * N + n0 + c];
  }
  __syncthreads();
#pragma unroll
  for (int q = 0; q < 4; ++q) {
    int r = r8 + q * 8;
    dst[(size_t)(n0 + r) * K + k0 + c] = f2bf(tile[c][r]);
  }
}

// bf16 WMMA GEMM with double-buffered ASYNC global->LDS staging.
// C(MxN,f32) = A(MxK, row-major) * B, B passed transposed BT[N][K].
__global__ __launch_bounds__(256) void k_gemm(const bf16_t* __restrict__ A,
                                              const bf16_t* __restrict__ BT,
                                              float* __restrict__ C,
                                              int M, int N, int K) {
  __shared__ bf16_t As[2][128 * 32];    // [m][k] double buffered
  __shared__ bf16_t BsT[2][128 * 32];   // [n][k] double buffered
  int tid  = threadIdx.x;
  int wave = tid >> 5, lane = tid & 31;
  int ll = lane & 15, hf = lane >> 4;
  int wr = wave >> 1, wc = wave & 1;          // 4x2 wave grid -> 32x64 per wave
  int rowBase = blockIdx.y * 128;
  int colBase = blockIdx.x * 128;
  bool interior = (colBase + 128 <= N);

  v8f acc[2][4];
#pragma unroll
  for (int i = 0; i < 2; ++i)
#pragma unroll
    for (int j = 0; j < 4; ++j)
#pragma unroll
      for (int e = 0; e < 8; ++e) acc[i][j][e] = 0.f;

  auto stage = [&](int kk, int sel) {
#pragma unroll
    for (int u = 0; u < 2; ++u) {
      int cch = tid + 256 * u;           // 0..511 chunks of 8 bf16
      int r   = cch >> 2;
      int c8  = (cch & 3) << 3;
      g2l_async_b128(lds_off(&As[sel][r * 32 + c8]),
                     A + (size_t)(rowBase + r) * K + kk + c8);
    }
    if (interior) {
#pragma unroll
      for (int u = 0; u < 2; ++u) {
        int cch = tid + 256 * u;
        int r   = cch >> 2;
        int c8  = (cch & 3) << 3;
        g2l_async_b128(lds_off(&BsT[sel][r * 32 + c8]),
                       BT + (size_t)(colBase + r) * K + kk + c8);
      }
    } else {
#pragma unroll
      for (int u = 0; u < 2; ++u) {
        int cch = tid + 256 * u;
        int r   = cch >> 2;
        int c8  = (cch & 3) << 3;
        int grow = colBase + r;
        uint4 val; val.x = val.y = val.z = val.w = 0u;
        if (grow < N) val = *(const uint4*)(BT + (size_t)grow * K + kk + c8);
        *(uint4*)(&BsT[sel][r * 32 + c8]) = val;
      }
    }
  };

  stage(0, 0);
  wait_async0();
  __syncthreads();

  int sel = 0;
  for (int kk = 0; kk < K; kk += 32) {
    if (kk + 32 < K) stage(kk + 32, sel ^ 1);   // async fill of next buffer
    if (kk + 64 < K) {                          // light prefetch two tiles out
      int r = tid >> 1, c16 = (tid & 1) << 4;
      __builtin_prefetch(A + (size_t)(rowBase + r) * K + kk + 64 + c16, 0, 3);
    }
    const bf16_t* cA = &As[sel][0];
    const bf16_t* cB = &BsT[sel][0];
    v16bf bfrag[4];
#pragma unroll
    for (int j = 0; j < 4; ++j)
      bfrag[j] = ldsBT(cB, 32, 0, wc * 64 + j * 16, lane);
#pragma unroll
    for (int i = 0; i < 2; ++i) {
      v16bf af = ldsA(cA, 32, wr * 32 + i * 16, 0, lane);
#pragma unroll
      for (int j = 0; j < 4; ++j)
        acc[i][j] = wmma_bf16(af, bfrag[j], acc[i][j]);
    }
    wait_async0();          // next tile landed
    __syncthreads();
    sel ^= 1;
  }
#pragma unroll
  for (int i = 0; i < 2; ++i)
#pragma unroll
    for (int j = 0; j < 4; ++j) {
      int col = colBase + wc * 64 + j * 16 + ll;
      if (col < N) {
#pragma unroll
        for (int e = 0; e < 8; ++e) {
          int row = rowBase + wr * 32 + i * 16 + e + 8 * hf;
          C[(size_t)row * N + col] = acc[i][j][e];
        }
      }
    }
}

// dt = clip(softplus(dt_raw+bias)); dA = dt*A; alpha = exp(A*dt); lam = sigmoid; dt_mean.
__global__ void k_dt(const float* __restrict__ proj, const float* __restrict__ A_log,
                     const float* __restrict__ dt_bias, float* __restrict__ dt_out,
                     float* __restrict__ dA, float* __restrict__ alpha,
                     float* __restrict__ lam, float* __restrict__ dtm) {
  int bt = blockIdx.x * blockDim.x + threadIdx.x;
  if (bt >= kM) return;
  const float* row = proj + (size_t)bt * kDProj + DTOFF;
  float s = 0.f;
  for (int h = 0; h < kNHeads; ++h) {
    float dtr = row[h] + dt_bias[h];
    float dt  = (dtr > 20.f) ? dtr : log1pf(__expf(dtr));
    dt = fminf(fmaxf(dt, 1e-4f), 0.5f);
    float A = -__expf(A_log[h]);
    size_t o = (size_t)bt * kNHeads + h;
    dt_out[o] = dt;
    dA[o]     = dt * A;
    alpha[o]  = __expf(A * dt);
    lam[o]    = 1.f / (1.f + __expf(-row[kNHeads + h]));
    s += dt;
  }
  dtm[bt] = s * (1.f / kNHeads);
}

// cum[b,t,ch] = -inclusive_cumsum_t(dt_mean * theta).
__global__ __launch_bounds__(256) void k_cum(const float* __restrict__ proj,
                                             const float* __restrict__ dtm,
                                             float* __restrict__ cum) {
  __shared__ float segsum[8][32];
  int b   = blockIdx.x;
  int ch  = threadIdx.x & 31;
  int seg = threadIdx.x >> 5;
  const int segT = kT / 8;
  size_t base = (size_t)b * kT;
  float s = 0.f;
  for (int t = seg * segT; t < (seg + 1) * segT; ++t)
    s += dtm[base + t] * proj[(base + t) * kDProj + THOFF + ch];
  segsum[seg][ch] = s;
  __syncthreads();
  float run = 0.f;
  for (int q = 0; q < seg; ++q) run += segsum[q][ch];
  for (int t = seg * segT; t < (seg + 1) * segT; ++t) {
    run += dtm[base + t] * proj[(base + t) * kDProj + THOFF + ch];
    cum[(base + t) * 32 + ch] = -run;
  }
}

// B/C rms-norm * bias + RoPE; store bf16 (C pre-scaled by kScale). 64 thr/block.
__global__ __launch_bounds__(64) void k_bc(const float* __restrict__ proj,
                                           const float* __restrict__ cum,
                                           const float* __restrict__ Bnb,
                                           const float* __restrict__ Cnb,
                                           bf16_t* __restrict__ Bbf,
                                           bf16_t* __restrict__ Cbf) {
  __shared__ float red[64];
  __shared__ float vb[64], vc[64];
  int bt = blockIdx.x;
  int i  = threadIdx.x;
  const float* row = proj + (size_t)bt * kDProj;
  float bval = row[BOFF + i];
  float cval = row[COFF + i];

  red[i] = bval * bval; __syncthreads();
  for (int s2 = 32; s2 > 0; s2 >>= 1) { if (i < s2) red[i] += red[i + s2]; __syncthreads(); }
  float rb = rsqrtf(red[0] * (1.f / 64.f) + kEps);
  __syncthreads();
  red[i] = cval * cval; __syncthreads();
  for (int s2 = 32; s2 > 0; s2 >>= 1) { if (i < s2) red[i] += red[i + s2]; __syncthreads(); }
  float rc = rsqrtf(red[0] * (1.f / 64.f) + kEps);
  __syncthreads();

  vb[i] = bval * rb * Bnb[i];
  vc[i] = cval * rc * Cnb[i];
  __syncthreads();

  if (i < 32) {
    float co = cosf(cum[(size_t)bt * 32 + i]);
    float si = sinf(cum[(size_t)bt * 32 + i]);
    float x1 = vb[i], x2 = vb[i + 32];
    Bbf[(size_t)bt * 64 + i]      = f2bf(x1 * co - x2 * si);
    Bbf[(size_t)bt * 64 + i + 32] = f2bf(x1 * si + x2 * co);
  } else {
    int q = i - 32;
    float co = cosf(cum[(size_t)bt * 32 + q]);
    float si = sinf(cum[(size_t)bt * 32 + q]);
    float x1 = vc[q], x2 = vc[q + 32];
    Cbf[(size_t)bt * 64 + q]      = f2bf((x1 * co - x2 * si) * kScale);
    Cbf[(size_t)bt * 64 + q + 32] = f2bf((x1 * si + x2 * co) * kScale);
  }
}

// x_gamma = silu(x)*lam*dt ; x_beta = silu(x[t-1])*(1-lam)*alpha*dt   (dt folded in)
__global__ void k_x(const float* __restrict__ proj, const float* __restrict__ dt,
                    const float* __restrict__ lam, const float* __restrict__ alpha,
                    bf16_t* __restrict__ Xg, bf16_t* __restrict__ Xb) {
  size_t idx = (size_t)blockIdx.x * blockDim.x + threadIdx.x;
  if (idx >= (size_t)kM * kDInner) return;
  int p  = (int)(idx & 63);
  int h  = (int)((idx >> 6) & 63);
  size_t bt = idx >> 12;
  int t = (int)(bt & (kT - 1));
  float xr = proj[bt * kDProj + XOFF + h * 64 + p];
  float sx = siluf(xr);
  size_t o = bt * kNHeads + h;
  float dtv = dt[o], la = lam[o];
  Xg[idx] = f2bf(sx * la * dtv);
  float xbv = 0.f;
  if (t > 0) {
    float xp = proj[(bt - 1) * kDProj + XOFF + h * 64 + p];
    xbv = siluf(xp) * (1.f - la) * alpha[o] * dtv;
  }
  Xb[idx] = f2bf(xbv);
}

// Per-(b,c,h) inclusive cumsum of dA over chunk; also chunk decay exp(Acs[last]).
__global__ __launch_bounds__(256) void k_acs(const float* __restrict__ dA,
                                             float* __restrict__ Acs,
                                             float* __restrict__ cdec) {
  __shared__ float s[kChunk];
  int blk = blockIdx.x;
  int h = blk % kNHeads;
  int c = (blk / kNHeads) % kNChunks;
  int b = blk / (kNHeads * kNChunks);
  int l = threadIdx.x;
  s[l] = dA[((size_t)b * kT + c * kChunk + l) * kNHeads + h];
  __syncthreads();
  for (int off = 1; off < kChunk; off <<= 1) {
    float add = (l >= off) ? s[l - off] : 0.f;
    __syncthreads();
    s[l] += add;
    __syncthreads();
  }
  Acs[(((size_t)b * kNChunks + c) * kNHeads + h) * kChunk + l] = s[l];
  if (l == kChunk - 1)
    cdec[((size_t)b * kNChunks + c) * kNHeads + h] = __expf(s[l]);
}

// Heavy SSD chunk kernel: one block per (b, chunk, head); 8 waves; dynamic LDS.
// C tile staged by the Tensor Data Mover; B tiles staged with async-to-LDS.
__global__ __launch_bounds__(256) void k_ssd(const bf16_t* __restrict__ Bbf,
                                             const bf16_t* __restrict__ Cbf,
                                             const bf16_t* __restrict__ Xg,
                                             const bf16_t* __restrict__ Xb,
                                             const float* __restrict__ Acs,
                                             float* __restrict__ Yout,
                                             float* __restrict__ St) {
  extern __shared__ char smem_raw[];
  bf16_t* sC   = (bf16_t*)smem_raw;          // [l=256][n=64]
  bf16_t* sB   = sC   + 256 * 64;            // [s=256][n=64]
  bf16_t* sBs  = sB   + 256 * 64;            // [s=256][n=64]
  bf16_t* sBT  = sBs  + 256 * 64;            // [n=64][l=256]
  bf16_t* sBsT = sBT  + 64 * 256;            // [n=64][l=256]
  bf16_t* sXgT = sBsT + 64 * 256;            // [p=64][l=256]
  bf16_t* sXbT = sXgT + 64 * 256;            // [p=64][l=256]
  bf16_t* sAtt = sXbT + 64 * 256;            // [l=256][s_local=64]
  float*  sAcs = (float*)(sAtt + 256 * 64);  // [256]
  float*  sDec = sAcs + 256;                 // [256]

  int blk = blockIdx.x;
  int h = blk % kNHeads;
  int c = (blk / kNHeads) % kNChunks;
  int b = blk / (kNHeads * kNChunks);
  int tid = threadIdx.x;
  int wave = tid >> 5, lane = tid & 31;
  int ll = lane & 15, hf = lane >> 4;
  int t0 = c * kChunk;

  // ---- stage: TDM for C, async-to-LDS for B / B_shift, regular for X ----
  if (wave == 0)
    tdm_load_1d(lds_off(sC), Cbf + ((size_t)b * kT + t0) * kDState, 256 * 64);
  {
    int l = tid;
    int t = t0 + l;
    size_t gBC = ((size_t)b * kT + t) * kDState;
#pragma unroll
    for (int q = 0; q < 8; ++q) {
      g2l_async_b128(lds_off(sB + l * 64 + q * 8), Bbf + gBC + q * 8);
      // shifted-B row (t-1); for the single t==0 row this reads the word
      // before Bbf (still inside the workspace) and is zero-fixed below.
      g2l_async_b128(lds_off(sBs + l * 64 + q * 8), Bbf + gBC - kDState + q * 8);
    }
    size_t gX = (((size_t)b * kT + t) * kNHeads + h) * kHeadDim;
    const uint4* gxg = (const uint4*)(Xg + gX);
    const uint4* gxb = (const uint4*)(Xb + gX);
#pragma unroll
    for (int q = 0; q < 8; ++q) {
      union { uint4 u; bf16_t hh[8]; } wg; wg.u = gxg[q];
      union { uint4 u; bf16_t hh[8]; } wb; wb.u = gxb[q];
#pragma unroll
      for (int e = 0; e < 8; ++e) {
        sXgT[(q * 8 + e) * 256 + l] = wg.hh[e];
        sXbT[(q * 8 + e) * 256 + l] = wb.hh[e];
      }
    }
    sAcs[l] = Acs[(((size_t)b * kNChunks + c) * kNHeads + h) * kChunk + l];
  }
  wait_async0();
  wait_tensor0();
  __syncthreads();

  sDec[tid] = __expf(sAcs[kChunk - 1] - sAcs[tid]);
  if (c == 0 && tid == 0) {     // zero the t==0 shifted-B row
    uint4 z; z.x = z.y = z.z = z.w = 0u;
    uint4* dBs = (uint4*)sBs;
#pragma unroll
    for (int q = 0; q < 8; ++q) dBs[q] = z;
  }
  { // build transposed copies of B / B_shift from the staged rows
    int l = tid;
    const uint4* rB  = (const uint4*)(sB + l * 64);
    const uint4* rBs = (const uint4*)(sBs + l * 64);
#pragma unroll
    for (int q = 0; q < 8; ++q) {
      union { uint4 u; bf16_t hh[8]; } w; w.u = rB[q];
#pragma unroll
      for (int e = 0; e < 8; ++e) sBT[(q * 8 + e) * 256 + l] = w.hh[e];
      union { uint4 u; bf16_t hh[8]; } w2; w2.u = rBs[q];
#pragma unroll
      for (int e = 0; e < 8; ++e) sBsT[(q * 8 + e) * 256 + l] = w2.hh[e];
    }
  }
  __syncthreads();

  // ---- Y_diag: Y = (Lm o C Bh^T) Xg + (Lm o C Bs^T) Xb ----
  v8f yacc[2][4];
#pragma unroll
  for (int i = 0; i < 2; ++i)
#pragma unroll
    for (int j = 0; j < 4; ++j)
#pragma unroll
      for (int e = 0; e < 8; ++e) yacc[i][j][e] = 0.f;

  int lrow0 = wave * 32;  // wave owns l-rows [lrow0, lrow0+32)

  for (int pass = 0; pass < 2; ++pass) {
    const bf16_t* pB  = pass ? sBs  : sB;    // [s][n]
    const bf16_t* pXT = pass ? sXbT : sXgT;  // [p][l]
    for (int sS = 0; sS < 4; ++sS) {
      int s0 = sS * 64;
#pragma unroll
      for (int ii = 0; ii < 2; ++ii) {
        int lt = wave * 2 + ii;
        int lbase = lt * 16;
#pragma unroll
        for (int st = 0; st < 4; ++st) {
          int sbase = s0 + st * 16;
          uint4 packed;
          if (sbase > lbase + 15) {  // tile fully above diagonal
            packed.x = packed.y = packed.z = packed.w = 0u;
          } else {
            v8f a;
#pragma unroll
            for (int e = 0; e < 8; ++e) a[e] = 0.f;
            v16bf af = ldsA(pB, 64, sbase, 0, lane);      // A(m=s,k=n)=B[s][n]
            v16bf bv = ldsBT(sC, 64, 0, lbase, lane);     // B(k=n,N=l)=C[l][n]
            a = wmma_bf16(af, bv, a);
            af = ldsA(pB, 64, sbase, 32, lane);
            bv = ldsBT(sC, 64, 32, lbase, lane);
            a = wmma_bf16(af, bv, a);
            int lcol = lbase + ll;
            float acl = sAcs[lcol];
            uint32_t w2[4];
#pragma unroll
            for (int d = 0; d < 4; ++d) {
              int srow = sbase + 8 * hf + 2 * d;
              float v0 = (srow     <= lcol) ? a[2*d]   * __expf(acl - sAcs[srow])     : 0.f;
              float v1 = (srow + 1 <= lcol) ? a[2*d+1] * __expf(acl - sAcs[srow + 1]) : 0.f;
              w2[d] = pack2(v0, v1);
            }
            packed.x = w2[0]; packed.y = w2[1]; packed.z = w2[2]; packed.w = w2[3];
          }
          *(uint4*)(sAtt + (size_t)(lbase + ll) * 64 + st * 16 + 8 * hf) = packed;
        }
      }
      __syncthreads();
#pragma unroll
      for (int i = 0; i < 2; ++i) {
        int lmax = lrow0 + i * 16 + 15;
#pragma unroll
        for (int kq = 0; kq < 2; ++kq) {
          if (s0 + kq * 32 > lmax) continue;  // strip entirely masked
          v16bf af = ldsA(sAtt, 64, lrow0 + i * 16, kq * 32, lane);
#pragma unroll
          for (int j = 0; j < 4; ++j) {
            v16bf bv = ldsBT(pXT, 256, s0 + kq * 32, j * 16, lane);
            yacc[i][j] = wmma_bf16(af, bv, yacc[i][j]);
          }
        }
      }
      __syncthreads();
    }
  }
#pragma unroll
  for (int i = 0; i < 2; ++i)
#pragma unroll
    for (int j = 0; j < 4; ++j)
#pragma unroll
      for (int e = 0; e < 8; ++e) {
        int lrow = lrow0 + i * 16 + e + 8 * hf;
        int p = j * 16 + ll;
        Yout[(((size_t)b * kT + t0 + lrow) * kNHeads + h) * kHeadDim + p] =
            yacc[i][j][e];
      }

  // ---- chunk states (n x p) = Bh^T (decay o Xg) + Bs^T (decay o Xb) ----
  v8f sacc[2];
#pragma unroll
  for (int u = 0; u < 2; ++u)
#pragma unroll
    for (int e = 0; e < 8; ++e) sacc[u][e] = 0.f;

  for (int pass = 0; pass < 2; ++pass) {
    const bf16_t* pBT = pass ? sBsT : sBT;   // [n][l]
    const bf16_t* pXT = pass ? sXbT : sXgT;  // [p][l]
#pragma unroll
    for (int u = 0; u < 2; ++u) {
      int tI = wave * 2 + u;
      int nt = tI >> 2, pt = tI & 3;
      for (int kq = 0; kq < 8; ++kq) {
        int l0k = kq * 32;
        v16bf af = ldsA(pBT, 256, nt * 16, l0k, lane);  // A(m=n,k=l)
        union { uint4 u4[2]; v16bf b; } raw;
        const char* bp =
            (const char*)(pXT + (size_t)(pt * 16 + ll) * 256 + l0k + 16 * hf);
        raw.u4[0] = *(const uint4*)(bp);
        raw.u4[1] = *(const uint4*)(bp + 16);
        v16bf bv;
#pragma unroll
        for (int e = 0; e < 16; ++e)
          bv[e] = f2bf(bf2f(raw.b[e]) * sDec[l0k + 16 * hf + e]);
        sacc[u] = wmma_bf16(af, bv, sacc[u]);
      }
    }
  }
#pragma unroll
  for (int u = 0; u < 2; ++u) {
    int tI = wave * 2 + u;
    int nt = tI >> 2, pt = tI & 3;
#pragma unroll
    for (int e = 0; e < 8; ++e) {
      int n = nt * 16 + e + 8 * hf;
      int p = pt * 16 + ll;
      St[((((size_t)b * kNChunks + c) * kNHeads + h) * kDState + n) * kHeadDim + p] =
          sacc[u][e];
    }
  }
}

// Inter-chunk linear scan: Prev[c] = carry ; carry = carry*cdec[c] + St[c].
__global__ __launch_bounds__(256) void k_scan(const float* __restrict__ St,
                                              const float* __restrict__ cdec,
                                              float* __restrict__ Prev) {
  int blk = blockIdx.x;  // b * kNHeads
  int h = blk % kNHeads;
  int b = blk / kNHeads;
  int tid = threadIdx.x;
  float carry[16];
#pragma unroll
  for (int e = 0; e < 16; ++e) carry[e] = 0.f;
  for (int c = 0; c < kNChunks; ++c) {
    size_t base = (((size_t)b * kNChunks + c) * kNHeads + h) * (size_t)(kDState * kHeadDim);
    float dec = cdec[((size_t)b * kNChunks + c) * kNHeads + h];
#pragma unroll
    for (int e = 0; e < 16; ++e) {
      int idx = tid + 256 * e;
      Prev[base + idx] = carry[e];
      carry[e] = carry[e] * dec + St[base + idx];
    }
  }
}

// Y += (Ch o exp(Acs)) @ Prev     (256x64 += 256x64 @ 64x64)
__global__ __launch_bounds__(256) void k_yoff(const bf16_t* __restrict__ Cbf,
                                              const float* __restrict__ Acs,
                                              const float* __restrict__ Prev,
                                              float* __restrict__ Y) {
  __shared__ bf16_t sA[256 * 64];  // 32 KB, [l][n]
  __shared__ bf16_t sPT[64 * 64];  // 8 KB,  [p][n]  (transposed Prev)
  int blk = blockIdx.x;
  int h = blk % kNHeads;
  int c = (blk / kNHeads) % kNChunks;
  int b = blk / (kNHeads * kNChunks);
  int tid = threadIdx.x;
  int wave = tid >> 5, lane = tid & 31;
  int ll = lane & 15, hf = lane >> 4;

  {
    int l = tid;
    float ea = __expf(Acs[(((size_t)b * kNChunks + c) * kNHeads + h) * kChunk + l]);
    const uint4* gC4 = (const uint4*)(Cbf + ((size_t)b * kT + c * kChunk + l) * kDState);
    uint4* dA4 = (uint4*)(sA + l * 64);
#pragma unroll
    for (int q = 0; q < 8; ++q) {
      union { uint4 u; bf16_t hh[8]; } w; w.u = gC4[q];
      uint4 o;
      o.x = pack2(bf2f(w.hh[0]) * ea, bf2f(w.hh[1]) * ea);
      o.y = pack2(bf2f(w.hh[2]) * ea, bf2f(w.hh[3]) * ea);
      o.z = pack2(bf2f(w.hh[4]) * ea, bf2f(w.hh[5]) * ea);
      o.w = pack2(bf2f(w.hh[6]) * ea, bf2f(w.hh[7]) * ea);
      dA4[q] = o;
    }
    const float* gP = Prev + (((size_t)b * kNChunks + c) * kNHeads + h) *
                                 (size_t)(kDState * kHeadDim);
#pragma unroll
    for (int q = 0; q < 16; ++q) {
      int idx = tid + 256 * q;  // = n*64 + p
      sPT[(idx & 63) * 64 + (idx >> 6)] = f2bf(gP[idx]);
    }
  }
  __syncthreads();

  v8f acc[2][4];
#pragma unroll
  for (int i = 0; i < 2; ++i)
#pragma unroll
    for (int j = 0; j < 4; ++j)
#pragma unroll
      for (int e = 0; e < 8; ++e) acc[i][j][e] = 0.f;

  int lrow0 = wave * 32;
#pragma unroll
  for (int i = 0; i < 2; ++i)
#pragma unroll
    for (int kq = 0; kq < 2; ++kq) {
      v16bf af = ldsA(sA, 64, lrow0 + i * 16, kq * 32, lane);
#pragma unroll
      for (int j = 0; j < 4; ++j) {
        v16bf bv = ldsBT(sPT, 64, kq * 32, j * 16, lane);
        acc[i][j] = wmma_bf16(af, bv, acc[i][j]);
      }
    }
#pragma unroll
  for (int i = 0; i < 2; ++i)
#pragma unroll
    for (int j = 0; j < 4; ++j)
#pragma unroll
      for (int e = 0; e < 8; ++e) {
        int lrow = lrow0 + i * 16 + e + 8 * hf;
        int p = j * 16 + ll;
        size_t idx = (((size_t)b * kT + c * kChunk + lrow) * kNHeads + h) * kHeadDim + p;
        Y[idx] += acc[i][j][e];
      }
}

// Final: y = rms_norm(Y_row) * silu(z), stored bf16 for the output GEMM.
__global__ __launch_bounds__(256) void k_norm(const float* __restrict__ Y,
                                              const float* __restrict__ proj,
                                              bf16_t* __restrict__ Ybf) {
  __shared__ float red[256];
  int bt = blockIdx.x;
  const float* y = Y + (size_t)bt * kDInner;
  const float* z = proj + (size_t)bt * kDProj + ZOFF;
  float ss = 0.f;
  for (int i = threadIdx.x; i < kDInner; i += 256) { float v = y[i]; ss += v * v; }
  red[threadIdx.x] = ss;
  __syncthreads();
  for (int s2 = 128; s2 > 0; s2 >>= 1) {
    if (threadIdx.x < s2) red[threadIdx.x] += red[threadIdx.x + s2];
    __syncthreads();
  }
  float r = rsqrtf(red[0] * (1.f / kDInner) + kEps);
  for (int i = threadIdx.x; i < kDInner; i += 256) {
    float g = siluf(z[i]);
    Ybf[(size_t)bt * kDInner + i] = f2bf(y[i] * r * g);
  }
}

// ------------------------- workspace layout -------------------------
constexpr size_t alignup(size_t x) { return (x + 255) & ~(size_t)255; }
constexpr size_t OFF_XBF    = 0;
constexpr size_t OFF_WINBF  = alignup(OFF_XBF    + (size_t)kM * kDModel * 2);
constexpr size_t OFF_PROJ   = alignup(OFF_WINBF  + (size_t)kDModel * kDProj * 2);
constexpr size_t OFF_DT     = alignup(OFF_PROJ   + (size_t)kM * kDProj * 4);
constexpr size_t OFF_DA     = alignup(OFF_DT     + (size_t)kM * kNHeads * 4);
constexpr size_t OFF_ALPHA  = alignup(OFF_DA     + (size_t)kM * kNHeads * 4);
constexpr size_t OFF_LAM    = alignup(OFF_ALPHA  + (size_t)kM * kNHeads * 4);
constexpr size_t OFF_DTM    = alignup(OFF_LAM    + (size_t)kM * kNHeads * 4);
constexpr size_t OFF_CUM    = alignup(OFF_DTM    + (size_t)kM * 4);
constexpr size_t OFF_BBF    = alignup(OFF_CUM    + (size_t)kM * 32 * 4);
constexpr size_t OFF_CBF    = alignup(OFF_BBF    + (size_t)kM * kDState * 2);
constexpr size_t OFF_XG     = alignup(OFF_CBF    + (size_t)kM * kDState * 2);
constexpr size_t OFF_XB     = alignup(OFF_XG     + (size_t)kM * kDInner * 2);
constexpr size_t OFF_ACS    = alignup(OFF_XB     + (size_t)kM * kDInner * 2);
constexpr size_t OFF_CDEC   = alignup(OFF_ACS    + (size_t)kBatch * kNChunks * kNHeads * kChunk * 4);
constexpr size_t OFF_ST     = alignup(OFF_CDEC   + (size_t)kBatch * kNChunks * kNHeads * 4);
constexpr size_t OFF_PREV   = alignup(OFF_ST     + (size_t)kBatch * kNChunks * kNHeads * kDState * kHeadDim * 4);
constexpr size_t OFF_Y      = alignup(OFF_PREV   + (size_t)kBatch * kNChunks * kNHeads * kDState * kHeadDim * 4);
constexpr size_t OFF_YBF    = alignup(OFF_Y      + (size_t)kM * kDInner * 4);
constexpr size_t OFF_WOUTBF = alignup(OFF_YBF    + (size_t)kM * kDInner * 2);

constexpr size_t SSD_LDS_BYTES = (size_t)8 * 256 * 64 * 2 + (size_t)2 * 256 * 4; // 264192

static inline int cdiv(int a, int b) { return (a + b - 1) / b; }

// ------------------------- launch -------------------------
extern "C" void kernel_launch(void* const* d_in, const int* in_sizes, int n_in,
                              void* d_out, int out_size, void* d_ws, size_t ws_size,
                              hipStream_t stream) {
  (void)in_sizes; (void)n_in; (void)out_size; (void)ws_size;
  const float* x       = (const float*)d_in[0];
  const float* W_in    = (const float*)d_in[1];
  const float* A_log   = (const float*)d_in[2];
  const float* dt_bias = (const float*)d_in[3];
  const float* Bnb     = (const float*)d_in[4];
  const float* Cnb     = (const float*)d_in[5];
  const float* W_out   = (const float*)d_in[6];
  float* out = (float*)d_out;

  char* ws = (char*)d_ws;
  bf16_t* xbf    = (bf16_t*)(ws + OFF_XBF);
  bf16_t* winbfT = (bf16_t*)(ws + OFF_WINBF);   // [kDProj][kDModel]
  float*  proj   = (float*)(ws + OFF_PROJ);
  float*  dt     = (float*)(ws + OFF_DT);
  float*  dA     = (float*)(ws + OFF_DA);
  float*  alpha  = (float*)(ws + OFF_ALPHA);
  float*  lam    = (float*)(ws + OFF_LAM);
  float*  dtm    = (float*)(ws + OFF_DTM);
  float*  cum    = (float*)(ws + OFF_CUM);
  bf16_t* Bbf    = (bf16_t*)(ws + OFF_BBF);
  bf16_t* Cbf    = (bf16_t*)(ws + OFF_CBF);
  bf16_t* Xg     = (bf16_t*)(ws + OFF_XG);
  bf16_t* Xb     = (bf16_t*)(ws + OFF_XB);
  float*  Acs    = (float*)(ws + OFF_ACS);
  float*  cdec   = (float*)(ws + OFF_CDEC);
  float*  St     = (float*)(ws + OFF_ST);
  float*  Prev   = (float*)(ws + OFF_PREV);
  float*  Y      = (float*)(ws + OFF_Y);
  bf16_t* Ybf    = (bf16_t*)(ws + OFF_YBF);
  bf16_t* woutbfT= (bf16_t*)(ws + OFF_WOUTBF);  // [kDModel][kDInner]

  (void)hipFuncSetAttribute((const void*)k_ssd,
                            hipFuncAttributeMaxDynamicSharedMemorySize,
                            (int)SSD_LDS_BYTES);

  {
    size_t n = (size_t)kM * kDModel;
    k_f2bf<<<cdiv((int)n, 256), 256, 0, stream>>>(x, xbf, n);
  }
  k_f2bfT<<<dim3(kDProj / 32, kDModel / 32), 256, 0, stream>>>(W_in, winbfT,
                                                               kDModel, kDProj);
  k_f2bfT<<<dim3(kDModel / 32, kDInner / 32), 256, 0, stream>>>(W_out, woutbfT,
                                                                kDInner, kDModel);
  k_gemm<<<dim3(cdiv(kDProj, 128), kM / 128), 256, 0, stream>>>(
      xbf, winbfT, proj, kM, kDProj, kDModel);
  k_dt<<<cdiv(kM, 256), 256, 0, stream>>>(proj, A_log, dt_bias, dt, dA, alpha, lam, dtm);
  k_cum<<<kBatch, 256, 0, stream>>>(proj, dtm, cum);
  k_bc<<<kM, 64, 0, stream>>>(proj, cum, Bnb, Cnb, Bbf, Cbf);
  {
    size_t n = (size_t)kM * kDInner;
    k_x<<<cdiv((int)n, 256), 256, 0, stream>>>(proj, dt, lam, alpha, Xg, Xb);
  }
  k_acs<<<kBatch * kNChunks * kNHeads, kChunk, 0, stream>>>(dA, Acs, cdec);
  k_ssd<<<kBatch * kNChunks * kNHeads, 256, SSD_LDS_BYTES, stream>>>(
      Bbf, Cbf, Xg, Xb, Acs, Y, St);
  k_scan<<<kBatch * kNHeads, 256, 0, stream>>>(St, cdec, Prev);
  k_yoff<<<kBatch * kNChunks * kNHeads, 256, 0, stream>>>(Cbf, Acs, Prev, Y);
  k_norm<<<kM, 256, 0, stream>>>(Y, proj, Ybf);
  k_gemm<<<dim3(cdiv(kDModel, 128), kM / 128), 256, 0, stream>>>(
      Ybf, woutbfT, out, kM, kDModel, kDInner);
}
